// TransLayer_8143257993458
// MI455X (gfx1250) — compile-verified, matrix operand order checked by reading, not run on hardware
//
#include <hip/hip_runtime.h>
#include <cstdint>

typedef __attribute__((ext_vector_type(16))) _Float16 v16h;
typedef __attribute__((ext_vector_type(8)))  _Float16 v8h;
typedef __attribute__((ext_vector_type(8)))  float    v8f;

constexpr int DIMC  = 512;
constexpr int HEADS = 8;
constexpr int DH    = 64;
constexpr int NLM   = 256;     // landmarks
constexpr int SEQ   = 8192;
constexpr int BATCH = 2;
constexpr int ROWS  = BATCH * SEQ;   // 16384
constexpr int KSZ   = 33;
constexpr float LN_EPS = 1e-5f;
constexpr float QSCALE = 0.125f;     // DH^-0.5

// ---------------------------------------------------------------------------
// B-operand fragment: B stored row-major [N,K]; fragment mirrors the A layout
// (lanes 0-15 hold n, K = k0+{0..7,16..23}; lanes 16-31 K = k0+{8..15,24..31}),
// so all loads are contiguous 16B (global_load_b128).
__device__ __forceinline__ v16h load_frag_nk(const _Float16* __restrict__ p, int ld,
                                             int n0, int k0) {
  const int lane = threadIdx.x & 31;
  const _Float16* q = p + (size_t)(n0 + (lane & 15)) * ld + (k0 + ((lane >> 4) << 3));
  v16h f;
#pragma unroll
  for (int e = 0; e < 8; ++e) f[e] = q[e];
#pragma unroll
  for (int e = 0; e < 8; ++e) f[e + 8] = q[e + 16];
  return f;
}

// ---------------------------------------------------------------------------
// WMMA GEMM: C = alpha*(A @ B^T) (+bias)(+resid)
// A f16 row-major [M,K]; B f16 row-major [N,K]. One wave per 16x16 C tile.
// A tiles are double-buffered through wave-private LDS with
// global_load_async_to_lds_b128; async loads complete in order, so the
// steady-state loop is unconditionally: issue-next -> s_wait_asynccnt 2 ->
// ds_load_b128 frag + global_load_b128 B frag -> wmma, with the last k-step
// peeled (s_wait_asynccnt 0). STORE_T stores C^T as one contiguous vector per
// lane for chaining results as B operands.
template <bool STORE_T, bool HAS_BIAS, bool HAS_RES, typename OutT>
__global__ __launch_bounds__(256) void gemm_wmma(
    const _Float16* __restrict__ A, const _Float16* __restrict__ B,
    OutT* __restrict__ C, int M, int N, int K, int lda, int ldb, int ldc,
    float alpha, const float* __restrict__ bias,
    const float* __restrict__ resid, int ldr) {
  __shared__ _Float16 stage[8][2][16 * 32];   // 2KB per wave, double-buffered
  const int lane  = threadIdx.x & 31;
  const int wave  = threadIdx.x >> 5;
  const int tilesX = N >> 4;
  const int tile  = blockIdx.x * 8 + wave;
  if (tile >= tilesX * (M >> 4)) return;
  const int row0 = (tile / tilesX) << 4;
  const int col0 = (tile % tilesX) << 4;

  // two 16B chunks per lane cover one 16x32-half A tile
  const int q0r = lane >> 2,       q0c = (lane & 3) * 8;
  const int q1r = (lane >> 2) + 8, q1c = q0c;
  unsigned d0[2], d1[2];
#pragma unroll
  for (int bu = 0; bu < 2; ++bu) {
    _Float16* st = &stage[wave][bu][0];
    d0[bu] = (unsigned)(uintptr_t)(st + q0r * 32 + q0c);
    d1[bu] = (unsigned)(uintptr_t)(st + q1r * 32 + q1c);
  }
  const int aoff = (lane & 15) * 32 + ((lane >> 4) << 3);
  const _Float16* arow0 = A + (size_t)row0 * lda;

  auto issue = [&](int bu, int k0) {
    const _Float16* g0 = arow0 + (size_t)q0r * lda + k0 + q0c;
    const _Float16* g1 = arow0 + (size_t)q1r * lda + k0 + q1c;
    asm volatile("global_load_async_to_lds_b128 %0, %1, off"
                 :: "v"(d0[bu]), "v"(g0) : "memory");
    asm volatile("global_load_async_to_lds_b128 %0, %1, off"
                 :: "v"(d1[bu]), "v"(g1) : "memory");
  };

  v8f acc = {};
  auto compute = [&](int bu, int k0) {
    // reference the __shared__ array directly so clang keeps addrspace(3)
    // and emits ds_load_b128 (buffer select folds into an immediate offset)
    v16h a;
    {
      v8h lo = *(const v8h*)(&stage[wave][bu][aoff]);
      v8h hi = *(const v8h*)(&stage[wave][bu][aoff + 16]);
#pragma unroll
      for (int e = 0; e < 8; ++e) { a[e] = lo[e]; a[e + 8] = hi[e]; }
    }
    v16h b = load_frag_nk(B, ldb, col0, k0);
    acc = __builtin_amdgcn_wmma_f32_16x16x32_f16(
        false, a, false, b, (short)0, acc, false, false);
  };

  issue(0, 0);                          // prime the pipeline
  int k0 = 0;
  for (; k0 + 32 < K; k0 += 32) {       // steady state: branch-free body
    const int cur = (k0 >> 5) & 1;
    issue(cur ^ 1, k0 + 32);            // prefetch next tile into other buffer
    asm volatile("s_wait_asynccnt 2" ::: "memory");  // current tile retired
    compute(cur, k0);
  }
  asm volatile("s_wait_asynccnt 0" ::: "memory");    // peeled last k-step
  compute((k0 >> 5) & 1, k0);

  const int cn    = col0 + (lane & 15);
  const int rbase = row0 + ((lane >> 4) << 3);
  float vals[8];
  float bv = 0.f;
  if constexpr (HAS_BIAS) bv = bias[cn];
#pragma unroll
  for (int r = 0; r < 8; ++r) {
    float v = acc[r] * alpha;
    if constexpr (HAS_BIAS) v += bv;
    if constexpr (HAS_RES)  v += resid[(size_t)(rbase + r) * ldr + cn];
    vals[r] = v;
  }
  if constexpr (STORE_T) {
    // 8 contiguous elements per lane: single vector store (b128 for f16)
    OutT* dst = C + (size_t)cn * ldc + rbase;
    if constexpr (sizeof(OutT) == 2) {
      v8h pk;
#pragma unroll
      for (int r = 0; r < 8; ++r) pk[r] = (_Float16)vals[r];
      *(v8h*)dst = pk;
    } else {
      v8f pk;
#pragma unroll
      for (int r = 0; r < 8; ++r) pk[r] = vals[r];
      *(v8f*)dst = pk;
    }
  } else {
#pragma unroll
    for (int r = 0; r < 8; ++r)
      C[(size_t)(rbase + r) * ldc + cn] = (OutT)vals[r];
  }
}

// ---------------- LayerNorm (writes f32 residual copy + f16 GEMM copy) ------
__global__ __launch_bounds__(256) void ln_kernel(
    const float* __restrict__ x, const float* __restrict__ g,
    const float* __restrict__ bt, float* __restrict__ xn,
    _Float16* __restrict__ xnh) {
  __shared__ float red[256];
  const float* xr = x + (size_t)blockIdx.x * DIMC;
  const int t = threadIdx.x;
  float v0 = xr[t], v1 = xr[t + 256];
  red[t] = v0 + v1;
  __syncthreads();
  for (int s = 128; s > 0; s >>= 1) { if (t < s) red[t] += red[t + s]; __syncthreads(); }
  float mu = red[0] * (1.0f / DIMC);
  __syncthreads();
  float d0 = v0 - mu, d1 = v1 - mu;
  red[t] = d0 * d0 + d1 * d1;
  __syncthreads();
  for (int s = 128; s > 0; s >>= 1) { if (t < s) red[t] += red[t + s]; __syncthreads(); }
  float rstd = rsqrtf(red[0] * (1.0f / DIMC) + LN_EPS);
  float o0 = d0 * rstd * g[t] + bt[t];
  float o1 = d1 * rstd * g[t + 256] + bt[t + 256];
  float* xo = xn + (size_t)blockIdx.x * DIMC;
  xo[t] = o0; xo[t + 256] = o1;
  _Float16* xh = xnh + (size_t)blockIdx.x * DIMC;
  xh[t] = (_Float16)o0; xh[t + 256] = (_Float16)o1;
}

// ---------------- small utility kernels -------------------------------------
__global__ void f32_to_f16_kernel(const float* __restrict__ s,
                                  _Float16* __restrict__ d, int n) {
  int i = blockIdx.x * blockDim.x + threadIdx.x;
  if (i < n) d[i] = (_Float16)s[i];
}

// [R,C] f32 row-major -> [C,R] f16 (B-operand layout for the GEMM)
__global__ void trans_f32_to_f16_kernel(const float* __restrict__ s,
                                        _Float16* __restrict__ d, int R, int C) {
  int i = blockIdx.x * blockDim.x + threadIdx.x;
  if (i < R * C) {
    int r = i / C, c = i % C;
    d[(size_t)c * R + r] = (_Float16)s[i];
  }
}

// f16 [NLM,NLM] transpose
__global__ void transpose_h_kernel(const _Float16* __restrict__ s,
                                   _Float16* __restrict__ d) {
  int i = blockIdx.x * blockDim.x + threadIdx.x;   // NLM*NLM
  int r = i / NLM, c = i % NLM;
  d[(size_t)c * NLM + r] = s[i];
}

__global__ void split_heads_kernel(const _Float16* __restrict__ qkv,
                                   _Float16* __restrict__ q,
                                   _Float16* __restrict__ k,
                                   _Float16* __restrict__ v) {
  size_t i = (size_t)blockIdx.x * blockDim.x + threadIdx.x;  // ROWS * 3*DIMC
  int col = (int)(i % (3 * DIMC));
  size_t row = i / (3 * DIMC);
  int b = (int)(row / SEQ), n = (int)(row % SEQ);
  int which = col / DIMC;
  int c = col % DIMC;
  int h = c / DH, dd = c % DH;
  size_t sl = (size_t)(b * HEADS + h) * SEQ * DH;
  _Float16 val = qkv[i];
  if (which == 0)      q[sl + (size_t)n * DH + dd] = (_Float16)((float)val * QSCALE);
  else if (which == 1) k[sl + (size_t)n * DH + dd] = val;
  else                 v[sl + (size_t)dd * SEQ + n] = val;   // v stored [DH,SEQ]
}

// mean over contiguous groups of 32 rows: src [SEQ,DH] -> dst [NLM,DH]
__global__ void landmark_kernel(const _Float16* __restrict__ src,
                                _Float16* __restrict__ dst) {
  int i = blockIdx.x * blockDim.x + threadIdx.x;   // NLM*DH
  int m = i / DH, dd = i % DH;
  const _Float16* p = src + (size_t)m * 32 * DH + dd;
  float s = 0.f;
#pragma unroll 8
  for (int j = 0; j < 32; ++j) s += (float)p[(size_t)j * DH];
  dst[i] = (_Float16)(s * (1.0f / 32.0f));
}

// row softmax; optional f32 output, always f16 output
__global__ __launch_bounds__(256) void softmax_kernel(
    const float* __restrict__ S, float* __restrict__ outF,
    _Float16* __restrict__ outH, int cols) {
  __shared__ float red[256];
  const int t = threadIdx.x;
  const float* row = S + (size_t)blockIdx.x * cols;
  float m = -1e30f;
  for (int c = t; c < cols; c += 256) m = fmaxf(m, row[c]);
  red[t] = m; __syncthreads();
  for (int s = 128; s > 0; s >>= 1) { if (t < s) red[t] = fmaxf(red[t], red[t + s]); __syncthreads(); }
  m = red[0]; __syncthreads();
  float sum = 0.f;
  for (int c = t; c < cols; c += 256) sum += __expf(row[c] - m);
  red[t] = sum; __syncthreads();
  for (int s = 128; s > 0; s >>= 1) { if (t < s) red[t] += red[t + s]; __syncthreads(); }
  float inv = 1.0f / red[0];
  for (int c = t; c < cols; c += 256) {
    float e = __expf(row[c] - m) * inv;
    if (outF) outF[(size_t)blockIdx.x * cols + c] = e;
    outH[(size_t)blockIdx.x * cols + c] = (_Float16)e;
  }
}

// z0 = a^T / (max(rowsum|a|)*max(colsum|a|)); writes z0 and z0^T (f16)
__global__ __launch_bounds__(256) void pinv_init_kernel(
    const float* __restrict__ a, _Float16* __restrict__ z,
    _Float16* __restrict__ zT) {
  __shared__ float red[256];
  const int i = threadIdx.x;
  float rs = 0.f, cs = 0.f;
  for (int j = 0; j < NLM; ++j) {
    rs += fabsf(a[(size_t)i * NLM + j]);
    cs += fabsf(a[(size_t)j * NLM + i]);
  }
  red[i] = rs; __syncthreads();
  for (int s = 128; s > 0; s >>= 1) { if (i < s) red[i] = fmaxf(red[i], red[i + s]); __syncthreads(); }
  float maxrow = red[0]; __syncthreads();
  red[i] = cs; __syncthreads();
  for (int s = 128; s > 0; s >>= 1) { if (i < s) red[i] = fmaxf(red[i], red[i + s]); __syncthreads(); }
  float sc = 1.0f / (maxrow * red[0]);
  for (int j = 0; j < NLM; ++j) {
    z [(size_t)i * NLM + j] = (_Float16)(a[(size_t)j * NLM + i] * sc);
    zT[(size_t)i * NLM + j] = (_Float16)(a[(size_t)i * NLM + j] * sc);
  }
}

// dst = (c0*I + c1*src)^T  (identity symmetric => dst = c0*I + c1*src^T),
// written in B-operand [N,K] layout for the next chained GEMM.
__global__ void id_axpy_t_kernel(const _Float16* __restrict__ src,
                                 _Float16* __restrict__ dst, float c0, float c1) {
  int i = blockIdx.x * blockDim.x + threadIdx.x;   // NLM*NLM
  int r = i / NLM, c = i % NLM;
  dst[(size_t)c * NLM + r] = (_Float16)(c1 * (float)src[i] + ((r == c) ? c0 : 0.f));
}

// depthwise 33-tap conv over sequence on v^T [DH,SEQ], accumulated into head-out
__global__ void conv_res_kernel(const _Float16* __restrict__ vT,
                                const float* __restrict__ w,
                                float* __restrict__ out, int h) {
  int i = blockIdx.x * blockDim.x + threadIdx.x;   // DH*SEQ
  int dd = i / SEQ, n = i % SEQ;
  float acc = 0.f;
#pragma unroll
  for (int t = 0; t < KSZ; ++t) {
    int nn = n + t - KSZ / 2;
    if (nn >= 0 && nn < SEQ) acc += w[h * KSZ + t] * (float)vT[(size_t)dd * SEQ + nn];
  }
  out[(size_t)n * DIMC + dd] += acc;
}

// ---------------- orchestration ---------------------------------------------
extern "C" void kernel_launch(void* const* d_in, const int* in_sizes, int n_in,
                              void* d_out, int out_size, void* d_ws, size_t ws_size,
                              hipStream_t stream) {
  const float* x      = (const float*)d_in[0];
  const float* ln_g   = (const float*)d_in[1];
  const float* ln_b   = (const float*)d_in[2];
  const float* W_qkv  = (const float*)d_in[3];
  const float* W_out  = (const float*)d_in[4];
  const float* b_out  = (const float*)d_in[5];
  const float* conv_w = (const float*)d_in[6];
  float* out = (float*)d_out;

  char* ws = (char*)d_ws;
  size_t off = 0;
  auto alloc = [&](size_t bytes) -> void* {
    void* p = ws + off;
    off += (bytes + 255) & ~(size_t)255;
    return p;
  };

  float*    xn     = (float*)   alloc((size_t)ROWS * DIMC * 4);
  _Float16* xnh    = (_Float16*)alloc((size_t)ROWS * DIMC * 2);
  _Float16* wqkvT  = (_Float16*)alloc((size_t)DIMC * 3 * DIMC * 2);  // [1536,512]
  _Float16* woutT  = (_Float16*)alloc((size_t)DIMC * DIMC * 2);      // [512,512]
  _Float16* qh     = (_Float16*)alloc((size_t)ROWS * DIMC * 2);      // [b,h,n,d]
  _Float16* kh     = (_Float16*)alloc((size_t)ROWS * DIMC * 2);      // [b,h,n,d]
  _Float16* vTh    = (_Float16*)alloc((size_t)ROWS * DIMC * 2);      // [b,h,d,n]
  float*    hout   = (float*)   alloc((size_t)ROWS * DIMC * 4);
  _Float16* hoh    = (_Float16*)alloc((size_t)ROWS * DIMC * 2);
  _Float16* qkvh   = (_Float16*)alloc((size_t)ROWS * 3 * DIMC * 2);  // 48 MB

  // per-(b,h) scratch overlaid on qkvh (consumed before this scratch is used)
  char* sc = (char*)qkvh;
  size_t so = 0;
  auto salloc = [&](size_t bytes) -> void* {
    void* p = sc + so;
    so += (bytes + 255) & ~(size_t)255;
    return p;
  };
  float*    sim1f  = (float*)   salloc((size_t)SEQ * NLM * 4);
  float*    sim3f  = (float*)   salloc((size_t)NLM * SEQ * 4);
  float*    sim2f  = (float*)   salloc((size_t)NLM * NLM * 4);
  float*    a2f    = (float*)   salloc((size_t)NLM * NLM * 4);
  _Float16* attn1h = (_Float16*)salloc((size_t)SEQ * NLM * 2);
  _Float16* attn3h = (_Float16*)salloc((size_t)NLM * SEQ * 2);
  _Float16* a2h    = (_Float16*)salloc((size_t)NLM * NLM * 2);
  _Float16* qlh    = (_Float16*)salloc((size_t)NLM * DH * 2);
  _Float16* klh    = (_Float16*)salloc((size_t)NLM * DH * 2);
  _Float16* zh     = (_Float16*)salloc((size_t)NLM * NLM * 2);
  _Float16* znh    = (_Float16*)salloc((size_t)NLM * NLM * 2);
  _Float16* zth    = (_Float16*)salloc((size_t)NLM * NLM * 2);  // z^T (B operand)
  _Float16* azh    = (_Float16*)salloc((size_t)NLM * NLM * 2);
  _Float16* t1t    = (_Float16*)salloc((size_t)NLM * NLM * 2);
  _Float16* uth    = (_Float16*)salloc((size_t)NLM * NLM * 2);
  _Float16* t2h    = (_Float16*)salloc((size_t)NLM * NLM * 2);
  _Float16* kvT    = (_Float16*)salloc((size_t)DH * NLM * 2);   // [DH,NLM]
  _Float16* w2T    = (_Float16*)salloc((size_t)DH * NLM * 2);   // [DH,NLM]

  auto gblocks = [](int M, int N) { return ((M >> 4) * (N >> 4) + 7) / 8; };
  const int eb = (NLM * NLM) / 256;

  // 1) LayerNorm
  ln_kernel<<<ROWS, 256, 0, stream>>>(x, ln_g, ln_b, xn, xnh);

  // 2) weights -> f16, transposed into B-operand [N,K] layout
  trans_f32_to_f16_kernel<<<(DIMC * 3 * DIMC + 255) / 256, 256, 0, stream>>>(
      W_qkv, wqkvT, DIMC, 3 * DIMC);
  trans_f32_to_f16_kernel<<<(DIMC * DIMC + 255) / 256, 256, 0, stream>>>(
      W_out, woutT, DIMC, DIMC);

  // 3) QKV GEMM: [16384,512] @ [512,1536] (B given as [1536,512])
  gemm_wmma<false, false, false, _Float16><<<gblocks(ROWS, 3 * DIMC), 256, 0, stream>>>(
      xnh, wqkvT, qkvh, ROWS, 3 * DIMC, DIMC, DIMC, DIMC, 3 * DIMC,
      1.0f, nullptr, nullptr, 0);

  // 4) split heads (q scaled; v stored transposed per head-slice)
  split_heads_kernel<<<(ROWS * 3 * DIMC) / 256, 256, 0, stream>>>(qkvh, qh, kh, vTh);

  // 5) per-(batch, head) Nystrom attention
  for (int b = 0; b < BATCH; ++b) {
    for (int h = 0; h < HEADS; ++h) {
      const size_t slice = (size_t)(b * HEADS + h) * SEQ * DH;
      const _Float16* qs  = qh + slice;    // [SEQ,DH]
      const _Float16* ks  = kh + slice;    // [SEQ,DH]
      const _Float16* vTs = vTh + slice;   // [DH,SEQ]
      float* hout_s = hout + (size_t)b * SEQ * DIMC + h * DH;

      landmark_kernel<<<(NLM * DH) / 256, 256, 0, stream>>>(qs, qlh);
      landmark_kernel<<<(NLM * DH) / 256, 256, 0, stream>>>(ks, klh);

      // sim1 = q @ k_l^T ; sim2 = q_l @ k_l^T ; sim3 = q_l @ k^T  (all A@B^T)
      gemm_wmma<false, false, false, float><<<gblocks(SEQ, NLM), 256, 0, stream>>>(
          qs, klh, sim1f, SEQ, NLM, DH, DH, DH, NLM, 1.0f, nullptr, nullptr, 0);
      gemm_wmma<false, false, false, float><<<gblocks(NLM, NLM), 256, 0, stream>>>(
          qlh, klh, sim2f, NLM, NLM, DH, DH, DH, NLM, 1.0f, nullptr, nullptr, 0);
      gemm_wmma<false, false, false, float><<<gblocks(NLM, SEQ), 256, 0, stream>>>(
          qlh, ks, sim3f, NLM, SEQ, DH, DH, DH, SEQ, 1.0f, nullptr, nullptr, 0);

      softmax_kernel<<<SEQ, 256, 0, stream>>>(sim1f, nullptr, attn1h, NLM);
      softmax_kernel<<<NLM, 256, 0, stream>>>(sim2f, a2f, a2h, NLM);
      softmax_kernel<<<NLM, 256, 0, stream>>>(sim3f, nullptr, attn3h, SEQ);

      // Newton-Schulz pinv of attn2 (a = a2h row-major; B operands transposed)
      pinv_init_kernel<<<1, 256, 0, stream>>>(a2f, zh, zth);
      _Float16* zc = zh;
      _Float16* zn = znh;
      for (int it = 0; it < 6; ++it) {
        gemm_wmma<false, false, false, _Float16><<<gblocks(NLM, NLM), 256, 0, stream>>>(
            a2h, zth, azh, NLM, NLM, NLM, NLM, NLM, NLM, 1.0f, nullptr, nullptr, 0);
        id_axpy_t_kernel<<<eb, 256, 0, stream>>>(azh, t1t, 7.0f, -1.0f);
        gemm_wmma<false, false, false, _Float16><<<gblocks(NLM, NLM), 256, 0, stream>>>(
            azh, t1t, t2h, NLM, NLM, NLM, NLM, NLM, NLM, 1.0f, nullptr, nullptr, 0);
        id_axpy_t_kernel<<<eb, 256, 0, stream>>>(t2h, uth, 15.0f, -1.0f);
        gemm_wmma<false, false, false, _Float16><<<gblocks(NLM, NLM), 256, 0, stream>>>(
            azh, uth, t2h, NLM, NLM, NLM, NLM, NLM, NLM, 1.0f, nullptr, nullptr, 0);
        id_axpy_t_kernel<<<eb, 256, 0, stream>>>(t2h, uth, 13.0f, -1.0f);
        gemm_wmma<false, false, false, _Float16><<<gblocks(NLM, NLM), 256, 0, stream>>>(
            zc, uth, zn, NLM, NLM, NLM, NLM, NLM, NLM, 0.25f, nullptr, nullptr, 0);
        _Float16* tmp = zc; zc = zn; zn = tmp;
        transpose_h_kernel<<<eb, 256, 0, stream>>>(zc, zth);
      }

      // out = attn1 @ (z @ (attn3 @ v))  [re-associated; chained via STORE_T]
      gemm_wmma<true, false, false, _Float16><<<gblocks(NLM, DH), 256, 0, stream>>>(
          attn3h, vTs, kvT, NLM, DH, SEQ, SEQ, SEQ, NLM, 1.0f, nullptr, nullptr, 0);
      gemm_wmma<true, false, false, _Float16><<<gblocks(NLM, DH), 256, 0, stream>>>(
          zc, kvT, w2T, NLM, DH, NLM, NLM, NLM, NLM, 1.0f, nullptr, nullptr, 0);
      gemm_wmma<false, false, false, float><<<gblocks(SEQ, DH), 256, 0, stream>>>(
          attn1h, w2T, hout_s, SEQ, DH, NLM, NLM, NLM, DIMC, 1.0f, nullptr, nullptr, 0);

      // depthwise conv residual on v^T, accumulated into head-out slice
      conv_res_kernel<<<(SEQ * DH) / 256, 256, 0, stream>>>(vTs, conv_w, hout_s, h);
    }
  }

  // 6) final projection with fused bias + LayerNorm residual:
  //    out = hout @ W_out + b_out + xn
  f32_to_f16_kernel<<<(ROWS * DIMC) / 256, 256, 0, stream>>>(hout, hoh, ROWS * DIMC);
  gemm_wmma<false, true, true, float><<<gblocks(ROWS, DIMC), 256, 0, stream>>>(
      hoh, woutT, out, ROWS, DIMC, DIMC, DIMC, DIMC, DIMC,
      1.0f, b_out, xn, DIMC);

  (void)in_sizes; (void)n_in; (void)out_size; (void)ws_size;
}
